// AttentionLayer_23922967838874
// MI455X (gfx1250) — matrix operand order repaired and checked by reading, not compiled, not run on hardware
//
#include <hip/hip_runtime.h>
#include <hip/hip_bf16.h>
#include <stdint.h>

// ---- problem constants (match reference) ----
#define B_ 4
#define N_ 2048
#define D_ 512
#define H_ 8
#define E_ 64
#define EPS_ 1e-6f

typedef _Float16 h8   __attribute__((ext_vector_type(8)));
typedef _Float16 v16h __attribute__((ext_vector_type(16)));
typedef float    v8f  __attribute__((ext_vector_type(8)));

// ---- WMMA wrapper: D = A(16x32 f16) * B(32x16 f16) + C(16x16 f32) ----
__device__ __forceinline__ v8f wmma16(v16h a, v16h b, v8f c) {
    return __builtin_amdgcn_wmma_f32_16x16x32_f16(
        /*neg_a=*/false, a, /*neg_b=*/false, b,
        /*c_mod=*/(short)0, c, /*reuse_a=*/false, /*reuse_b=*/false);
}

// A-fragment: lane = row m=lane%16; element j -> k = (j<8?0:16)+8*hl+j%8.
__device__ __forceinline__ v16h load_a(const _Float16* rowp, int hl) {
    h8 lo = *(const h8*)(rowp + hl * 8);
    h8 hi = *(const h8*)(rowp + 16 + hl * 8);
    v16h r;
#pragma unroll
    for (int i = 0; i < 8; ++i) { r[i] = lo[i]; r[i + 8] = hi[i]; }
    return r;
}

// B-fragment: lane = column n=lane%16; element j -> k = (j<8?0:16)+8*hl+j%8 (row stride ld).
__device__ __forceinline__ v16h load_b(const _Float16* p, int ld, int hl) {
    v16h r;
#pragma unroll
    for (int i = 0; i < 8; ++i) { r[i] = p[(size_t)(8 * hl + i) * ld]; r[i + 8] = p[(size_t)(16 + 8 * hl + i) * ld]; }
    return r;
}

// Async global->LDS 16B copy (gfx1250, ASYNCcnt-tracked). GVS mode:
// saddr = uniform 64-bit base, vaddr = per-lane 32-bit byte offset, vdst = LDS byte addr.
__device__ __forceinline__ void async_b128(unsigned lds_addr, unsigned goff, uint64_t saddr) {
    asm volatile("global_load_async_to_lds_b128 %0, %1, %2"
                 :: "v"(lds_addr), "v"(goff), "s"(saddr) : "memory");
}
__device__ __forceinline__ void async_b128_off(unsigned lds_addr, unsigned goff, uint64_t saddr,
                                               int imm) {
    if (imm == 131072)
        asm volatile("global_load_async_to_lds_b128 %0, %1, %2 offset:131072"
                     :: "v"(lds_addr), "v"(goff), "s"(saddr) : "memory");
    else
        asm volatile("global_load_async_to_lds_b128 %0, %1, %2 offset:2048"
                     :: "v"(lds_addr), "v"(goff), "s"(saddr) : "memory");
}
__device__ __forceinline__ unsigned lds_u32(const void* p) {
    return (unsigned)(uintptr_t)p;   // flat addr low 32 bits == LDS byte offset
}

// ---------------- kernel 1: weights f32 -> f16 (layouts kept natural) ----------------
__global__ void prep_weights(const float* __restrict__ qp, const float* __restrict__ kp,
                             const float* __restrict__ vp, const float* __restrict__ gp,
                             const float* __restrict__ ow,
                             _Float16* __restrict__ wq, _Float16* __restrict__ wk,
                             _Float16* __restrict__ wv, _Float16* __restrict__ wg,
                             _Float16* __restrict__ owh) {
    int i = blockIdx.x * blockDim.x + threadIdx.x;
    const int PW = H_ * D_ * E_;          // 262144, == D_*D_ too
    if (i < PW) {
        wq[i] = (_Float16)qp[i];
        wk[i] = (_Float16)kp[i];
        wv[i] = (_Float16)vp[i];
        wg[i] = (_Float16)gp[i];
        owh[i] = (_Float16)ow[i];
    }
}

// ---------------- kernel 2: input LayerNorm, write f16 activations ----------------
__global__ void ln_in(const float* __restrict__ x, const float* __restrict__ gamma,
                      const float* __restrict__ beta, _Float16* __restrict__ xn) {
    __shared__ float s1[8], s2[8];
    size_t row = blockIdx.x;
    const float* xr = x + row * D_;
    float a = xr[threadIdx.x];
    float b = xr[threadIdx.x + 256];
    float s = a + b, ss = a * a + b * b;
#pragma unroll
    for (int m = 16; m; m >>= 1) { s += __shfl_xor(s, m, 32); ss += __shfl_xor(ss, m, 32); }
    int wv = threadIdx.x >> 5, ln = threadIdx.x & 31;
    if (!ln) { s1[wv] = s; s2[wv] = ss; }
    __syncthreads();
    if (threadIdx.x < 32) {
        s  = (threadIdx.x < 8) ? s1[threadIdx.x] : 0.f;
        ss = (threadIdx.x < 8) ? s2[threadIdx.x] : 0.f;
#pragma unroll
        for (int m = 4; m; m >>= 1) { s += __shfl_xor(s, m, 32); ss += __shfl_xor(ss, m, 32); }
        if (!threadIdx.x) { s1[0] = s; s2[0] = ss; }
    }
    __syncthreads();
    float mean = s1[0] * (1.f / D_);
    float var  = s2[0] * (1.f / D_) - mean * mean;
    float rstd = rsqrtf(var + EPS_);
    xn[row * D_ + threadIdx.x]       = (_Float16)(gamma[threadIdx.x] * ((a - mean) * rstd) + beta[threadIdx.x]);
    xn[row * D_ + threadIdx.x + 256] = (_Float16)(gamma[threadIdx.x + 256] * ((b - mean) * rstd) + beta[threadIdx.x + 256]);
}

// ---------------- kernel 3: Q/K/V/Gate projections + RoPE ----------------
__global__ void qkvg_proj(const _Float16* __restrict__ xn,
                          const _Float16* __restrict__ wq, const _Float16* __restrict__ wk,
                          const _Float16* __restrict__ wv, const _Float16* __restrict__ wg,
                          const float* __restrict__ rcos, const float* __restrict__ rsin,
                          _Float16* __restrict__ q16, _Float16* __restrict__ kt16,
                          _Float16* __restrict__ v16o, float* __restrict__ gate) {
    const int wid = threadIdx.x >> 5, lane = threadIdx.x & 31;
    const int hl = lane >> 4, lm = lane & 15;
    const int tn = blockIdx.x * 16, te = blockIdx.y * 16;
    const int bh = blockIdx.z, b = bh >> 3, h = bh & 7;

    const _Float16* W = (wid == 0) ? wq : (wid == 1) ? wk : (wid == 2) ? wv : wg;
    const _Float16* wbase = W + (size_t)h * D_ * E_ + te;                 // (D,E) row-major
    const _Float16* abase = xn + ((size_t)(b * N_) + tn + lm) * D_;       // (BN,D) row-major

    v8f c = {};
    for (int k0 = 0; k0 < D_; k0 += 32) {
        v16h a = load_a(abase + k0, hl);
        v16h bf = load_b(wbase + (size_t)k0 * E_ + lm, E_, hl);
        c = wmma16(a, bf, c);
    }

    if (wid < 2) {           // Q or K: RoPE then store
#pragma unroll
        for (int r = 0; r < 8; ++r) {
            int n = tn + r + 8 * hl;
            int e = te + lm;
            float xv = c[r];
            float partner = __shfl_xor(xv, 1, 32);    // neighbor column e^1
            float cs = rcos[n * E_ + e], sn = rsin[n * E_ + e];
            float res = (e & 1) ? (partner * sn + xv * cs) : (xv * cs - partner * sn);
            if (wid == 0) q16[((size_t)bh * N_ + n) * E_ + e] = (_Float16)res;
            else          kt16[((size_t)bh * E_ + e) * N_ + n] = (_Float16)res;
        }
    } else if (wid == 2) {   // V natural layout
#pragma unroll
        for (int r = 0; r < 8; ++r) {
            int n = tn + r + 8 * hl;
            v16o[((size_t)bh * N_ + n) * E_ + te + lm] = (_Float16)c[r];
        }
    } else {                 // gate = sigmoid
#pragma unroll
        for (int r = 0; r < 8; ++r) {
            int n = tn + r + 8 * hl;
            gate[((size_t)bh * N_ + n) * E_ + te + lm] = 1.f / (1.f + __expf(-c[r]));
        }
    }
}

// ---------------- kernel 4: flash attention + gating, async-LDS staged K/V ----------------
// block = 128 (4 waves, 4 query tiles of 16 rows, same b,h). Per 32-wide KV step the block
// stages K (64x32, 4KB) and V (32x64, 4KB) into double-buffered LDS via
// global_load_async_to_lds_b128 (4 per lane), overlapped with the previous tile's 8 WMMAs.
__global__ void attention(const _Float16* __restrict__ q16, const _Float16* __restrict__ kt16,
                          const _Float16* __restrict__ vmat, const float* __restrict__ gate,
                          const float* __restrict__ mask, _Float16* __restrict__ oh) {
    __shared__ _Float16 kbuf[2][64 * 32];
    __shared__ _Float16 vbuf[2][32 * 64];
    __shared__ _Float16 plds[4][16 * 32];
    const int wid = threadIdx.x >> 5, lane = threadIdx.x & 31;
    const int hl = lane >> 4, lm = lane & 15;
    const int bh = blockIdx.y, b = bh >> 3, h = bh & 7;
    const int tn = blockIdx.x * 64 + wid * 16;
    const float scale = 0.125f;                         // 1/sqrt(E)
    const float* mb = mask + (size_t)b * N_;

    // per-thread async-copy coordinates (block-wide cooperative copy)
    const unsigned t = threadIdx.x;
    const unsigned krow = t >> 2, kpart = t & 3;        // K: 64 rows x 4 x16B parts
    const unsigned kgoff = (krow * (unsigned)N_ + kpart * 8u) * 2u;
    const unsigned vrow = t >> 3, vpart = t & 7;        // V: 32 rows x 8 x16B parts
    const unsigned vgoff = (vrow * (unsigned)E_ + vpart * 8u) * 2u;
    const uint64_t kbase = (uint64_t)(uintptr_t)(kt16 + (size_t)bh * E_ * N_);
    const uint64_t vbase = (uint64_t)(uintptr_t)(vmat + (size_t)bh * N_ * E_);

    auto issue = [&](int kv, int pb) {
        uint64_t ks = kbase + (unsigned)kv * 2u;                    // +kv columns (bytes)
        uint64_t vs = vbase + (unsigned)(kv * E_) * 2u;             // +kv rows (bytes)
        async_b128(lds_u32(&kbuf[pb][krow * 32 + kpart * 8]), kgoff, ks);
        async_b128_off(lds_u32(&kbuf[pb][(krow + 32) * 32 + kpart * 8]), kgoff, ks, 131072);
        async_b128(lds_u32(&vbuf[pb][vrow * 64 + vpart * 8]), vgoff, vs);
        async_b128_off(lds_u32(&vbuf[pb][(vrow + 16) * 64 + vpart * 8]), vgoff, vs, 2048);
    };

    const _Float16* qbase = q16 + ((size_t)bh * N_ + tn + lm) * E_;
    v16h qa0 = load_a(qbase, hl);                       // k = e 0..31
    v16h qa1 = load_a(qbase + 32, hl);                  // k = e 32..63

    v8f o0 = {}, o1 = {}, o2 = {}, o3 = {};
    float mrow[8], lrow[8];
#pragma unroll
    for (int r = 0; r < 8; ++r) { mrow[r] = -3.0e38f; lrow[r] = 0.f; }

    issue(0, 0);
    const int T = N_ / 32;
    for (int it = 0; it < T; ++it) {
        const int kv = it * 32;
        const int pb = it & 1;
        __syncthreads();                          // buf[pb^1] readers (it-1) done before re-issue
        if (it + 1 < T) {
            issue(kv + 32, pb ^ 1);
            asm volatile("s_wait_asynccnt 0x4" ::: "memory");   // batch `it` landed, it+1 in flight
        } else {
            asm volatile("s_wait_asynccnt 0x0" ::: "memory");
        }
        __syncthreads();                          // all lanes' async writes visible block-wide

        const _Float16* kb0 = &kbuf[pb][lm];                 // rows e 0..31, column kv = lm
        const _Float16* kb1 = &kbuf[pb][32 * 32 + lm];       // rows e 32..63, column kv = lm
        v8f s = {}, s2 = {};
        s  = wmma16(qa0, load_b(kb0, 32, hl), s);
        s  = wmma16(qa1, load_b(kb1, 32, hl), s);
        s2 = wmma16(qa0, load_b(kb0 + 16, 32, hl), s2);
        s2 = wmma16(qa1, load_b(kb1 + 16, 32, hl), s2);

        float km_lo = (mb[kv + lm]      > 0.f) ? 1.f : 0.f;
        float km_hi = (mb[kv + 16 + lm] > 0.f) ? 1.f : 0.f;
        float ng_lo = km_lo > 0.f ? 0.f : -1.0e9f;
        float ng_hi = km_hi > 0.f ? 0.f : -1.0e9f;

#pragma unroll
        for (int r = 0; r < 8; ++r) {
            float slo = s[r] * scale + ng_lo;
            float shi = s2[r] * scale + ng_hi;
            float mx = fmaxf(slo, shi);
#pragma unroll
            for (int mk = 8; mk; mk >>= 1) mx = fmaxf(mx, __shfl_xor(mx, mk, 32));
            float mnew = fmaxf(mrow[r], mx);
            float alpha = __expf(mrow[r] - mnew);
            mrow[r] = mnew;
            float plo = __expf(slo - mnew) * km_lo;
            float phi = __expf(shi - mnew) * km_hi;
            float rs = plo + phi;
#pragma unroll
            for (int mk = 8; mk; mk >>= 1) rs += __shfl_xor(rs, mk, 32);
            lrow[r] = lrow[r] * alpha + rs;
            o0[r] *= alpha; o1[r] *= alpha; o2[r] *= alpha; o3[r] *= alpha;
            int rr = r + 8 * hl;
            plds[wid][rr * 32 + lm]      = (_Float16)plo;
            plds[wid][rr * 32 + lm + 16] = (_Float16)phi;
        }
        // P: C-layout -> A-layout via per-wave LDS staging (same-wave, counter waits only)
        v16h pa = load_a(&plds[wid][lm * 32], hl);
        const _Float16* vb = &vbuf[pb][lm];                  // column e = lm, rows kv-local 0..31
        o0 = wmma16(pa, load_b(vb, 64, hl),      o0);
        o1 = wmma16(pa, load_b(vb + 16, 64, hl), o1);
        o2 = wmma16(pa, load_b(vb + 32, 64, hl), o2);
        o3 = wmma16(pa, load_b(vb + 48, 64, hl), o3);
    }

#pragma unroll
    for (int r = 0; r < 8; ++r) {
        int n = tn + r + 8 * hl;
        float qm  = (mb[n] > 0.f) ? 1.f : 0.f;
        float rcp = qm / fmaxf(lrow[r], 1e-20f);
        const float* g = gate + ((size_t)bh * N_ + n) * E_;
        size_t ob = ((size_t)(b * N_ + n)) * (H_ * E_) + h * E_;
        oh[ob + lm]      = (_Float16)(o0[r] * rcp * g[lm]);
        oh[ob + 16 + lm] = (_Float16)(o1[r] * rcp * g[16 + lm]);
        oh[ob + 32 + lm] = (_Float16)(o2[r] * rcp * g[32 + lm]);
        oh[ob + 48 + lm] = (_Float16)(o3[r] * rcp * g[48 + lm]);
    }
}

// ---------------- kernel 5: out projection + bias + residual + LayerNorm ----------------
__global__ void out_proj(const _Float16* __restrict__ oh, const _Float16* __restrict__ ow,
                         const float* __restrict__ ob, const float* __restrict__ x,
                         const float* __restrict__ gamma, const float* __restrict__ beta,
                         float* __restrict__ out) {
    __shared__ float buf[16 * 512];
    const int wid = threadIdx.x >> 5, lane = threadIdx.x & 31;
    const int hl = lane >> 4, lm = lane & 15;
    const int tr = blockIdx.x * 16;
    const _Float16* abase = oh + (size_t)(tr + lm) * 512;

#pragma unroll
    for (int t = 0; t < 4; ++t) {
        int tc = (wid + t * 8) * 16;
        v8f c = {};
        for (int k0 = 0; k0 < 512; k0 += 32) {
            v16h a = load_a(abase + k0, hl);
            v16h bf = load_b(ow + (size_t)k0 * 512 + tc + lm, 512, hl);
            c = wmma16(a, bf, c);
        }
        int cc = tc + lm;
        float bias = ob[cc];
#pragma unroll
        for (int r = 0; r < 8; ++r) {
            int rr = r + 8 * hl;
            buf[rr * 512 + cc] = c[r] + bias + x[(size_t)(tr + rr) * 512 + cc];
        }
    }
    __syncthreads();

    int row = threadIdx.x >> 4;          // 16 threads per row
    int sub = threadIdx.x & 15;
    float s = 0.f, ss = 0.f;
#pragma unroll
    for (int i = 0; i < 32; ++i) {
        float v = buf[row * 512 + sub + i * 16];
        s += v; ss += v * v;
    }
#pragma unroll
    for (int mk = 8; mk; mk >>= 1) { s += __shfl_xor(s, mk, 32); ss += __shfl_xor(ss, mk, 32); }
    float mean = s * (1.f / 512.f);
    float var  = ss * (1.f / 512.f) - mean * mean;
    float rstd = rsqrtf(var + EPS_);
#pragma unroll
    for (int i = 0; i < 32; ++i) {
        int cidx = sub + i * 16;
        float v = buf[row * 512 + cidx];
        out[(size_t)(tr + row) * 512 + cidx] = gamma[cidx] * ((v - mean) * rstd) + beta[cidx];
    }
}

// ---------------- launch ----------------
extern "C" void kernel_launch(void* const* d_in, const int* in_sizes, int n_in,
                              void* d_out, int out_size, void* d_ws, size_t ws_size,
                              hipStream_t stream) {
    (void)in_sizes; (void)n_in; (void)out_size; (void)ws_size;
    const float* x      = (const float*)d_in[0];
    const float* mask   = (const float*)d_in[1];
    const float* q_proj = (const float*)d_in[2];
    const float* k_proj = (const float*)d_in[3];
    const float* v_proj = (const float*)d_in[4];
    const float* g_proj = (const float*)d_in[5];
    const float* gin    = (const float*)d_in[6];
    const float* bin    = (const float*)d_in[7];
    const float* gout   = (const float*)d_in[8];
    const float* bout   = (const float*)d_in[9];
    const float* out_w  = (const float*)d_in[10];
    const float* out_b  = (const float*)d_in[11];
    const float* rcos   = (const float*)d_in[12];
    const float* rsin   = (const float*)d_in[13];
    float* out = (float*)d_out;

    char* ws = (char*)d_ws;
    size_t off = 0;
    auto take = [&](size_t bytes) { char* p = ws + off; off += (bytes + 255) & ~(size_t)255; return p; };
    const size_t BN = (size_t)B_ * N_;
    const size_t BHNE = (size_t)B_ * H_ * N_ * E_;

    _Float16* xn   = (_Float16*)take(BN * D_ * 2);
    _Float16* wq16 = (_Float16*)take((size_t)H_ * D_ * E_ * 2);
    _Float16* wk16 = (_Float16*)take((size_t)H_ * D_ * E_ * 2);
    _Float16* wv16 = (_Float16*)take((size_t)H_ * D_ * E_ * 2);
    _Float16* wg16 = (_Float16*)take((size_t)H_ * D_ * E_ * 2);
    _Float16* ow16 = (_Float16*)take((size_t)512 * 512 * 2);
    _Float16* q16  = (_Float16*)take(BHNE * 2);
    _Float16* kt16 = (_Float16*)take(BHNE * 2);
    _Float16* v16b = (_Float16*)take(BHNE * 2);
    float*    gate = (float*)take(BHNE * 4);
    _Float16* oh   = (_Float16*)take(BN * 512 * 2);

    prep_weights<<<(H_ * D_ * E_ + 255) / 256, 256, 0, stream>>>(
        q_proj, k_proj, v_proj, g_proj, out_w, wq16, wk16, wv16, wg16, ow16);
    ln_in<<<(int)BN, 256, 0, stream>>>(x, gin, bin, xn);
    qkvg_proj<<<dim3(N_ / 16, E_ / 16, B_ * H_), 128, 0, stream>>>(
        xn, wq16, wk16, wv16, wg16, rcos, rsin, q16, kt16, v16b, gate);
    attention<<<dim3(N_ / 64, B_ * H_), 128, 0, stream>>>(
        q16, kt16, v16b, gate, mask, oh);
    out_proj<<<(int)(BN / 16), 256, 0, stream>>>(
        oh, ow16, out_b, x, gout, bout, out);
}
